// Decoder_55001351192866
// MI455X (gfx1250) — compile-verified, matrix operand order checked by reading, not run on hardware
//
#include <hip/hip_runtime.h>

// ---------------------------------------------------------------------------
// Problem constants (from reference): B=32, LX=64, T=64, D=512, DW=512, V=32000
// ---------------------------------------------------------------------------
#define BB   32
#define LX   64
#define TT   64
#define DD   512
#define VV   32000
#define KGATE 2048        // DW + 2D + D  (xh = [emb | input_feed | h])
#define NGATE 2048        // 4*D
#define KC2R  1536        // 3*D

typedef __attribute__((ext_vector_type(16))) __bf16 v16bf;
typedef __attribute__((ext_vector_type(8)))  float  v8f;

__device__ __forceinline__ unsigned short f2bf(float f) {
  unsigned int u = __float_as_uint(f);
  u = u + 0x7FFFu + ((u >> 16) & 1u);           // round-to-nearest-even
  return (unsigned short)(u >> 16);
}
__device__ __forceinline__ float sigmoidf_(float x) { return 1.0f / (1.0f + __expf(-x)); }

// ---------------------------------------------------------------------------
// Weight conversion kernels (fp32 -> bf16, operand layouts for WMMA)
// ---------------------------------------------------------------------------
// Wcat_nk[j][k], j in [0,2048): k<1536 -> W_ih[j][k], else W_hh[j][k-1536]
__global__ void cvt_wcat(const float* __restrict__ W_ih, const float* __restrict__ W_hh,
                         unsigned short* __restrict__ out) {
  int idx = blockIdx.x * 256 + threadIdx.x;
  if (idx >= NGATE * KGATE) return;
  int j = idx >> 11, k = idx & 2047;
  float v = (k < 1536) ? W_ih[(size_t)j * 1536 + k] : W_hh[(size_t)j * 512 + (k - 1536)];
  out[idx] = f2bf(v);
}
// direct copy-convert (already [N][K] row-major)
__global__ void cvt_direct(const float* __restrict__ in, unsigned short* __restrict__ out, int n) {
  int idx = blockIdx.x * 256 + threadIdx.x;
  if (idx < n) out[idx] = f2bf(in[idx]);
}
// in is [K][N] row-major ([in,out]); produce out_nk[n][k]
__global__ void cvt_transpose(const float* __restrict__ in, unsigned short* __restrict__ out,
                              int K, int N) {
  int idx = blockIdx.x * 256 + threadIdx.x;
  if (idx >= K * N) return;
  int k = idx / N, n = idx % N;
  out[(size_t)n * K + k] = f2bf(in[idx]);
}
__global__ void cvt_bias(const float* __restrict__ b_ih, const float* __restrict__ b_hh,
                         float* __restrict__ out, int n) {
  int idx = blockIdx.x * 256 + threadIdx.x;
  if (idx < n) out[idx] = b_ih[idx] + b_hh[idx];
}
// h := dec_init_h, c := dec_init_c, ctx := 0
__global__ void init_state(const float* __restrict__ h0, const float* __restrict__ c0,
                           float* __restrict__ h, float* __restrict__ c,
                           float* __restrict__ ctx) {
  int idx = blockIdx.x * 256 + threadIdx.x;           // 32*1024 threads
  if (idx < BB * DD) { h[idx] = h0[idx]; c[idx] = c0[idx]; }
  if (idx < BB * 2 * DD) ctx[idx] = 0.0f;
}

// ---------------------------------------------------------------------------
// WMMA GEMM, specialized for M == 32 (both 16-row tiles fused per wave so each
// B fragment is fetched exactly once) and N-blocked by 2 tiles per wave (4
// independent accumulators -> WMMA dependency distance 4, hides the bf16
// WMMA->WMMA RAW latency). A row-major bf16 [32][K]; B supplied as Bt
// row-major [N][K] so each lane's column is two contiguous 16B runs.
// blockDim = 256 (8 waves, wave32); requires N % 32 == 0, K % 32 == 0.
// Epilogue: +bias[n], optional tanh, optional bf16 copy of C.
// ---------------------------------------------------------------------------
__device__ __forceinline__ v16bf load_frag(const unsigned short* __restrict__ base,
                                           int row16, int ldk, int k0, int lane) {
  // 16-bit operand layout: lanes 0-15 => K = 0..7 / 16..23 ; lanes 16-31 => +8
  const unsigned short* p = base + (size_t)(row16 + (lane & 15)) * ldk + k0 + ((lane >> 4) << 3);
  const __bf16* q0 = (const __bf16*)p;
  const __bf16* q1 = (const __bf16*)(p + 16);
  v16bf r;
#pragma unroll
  for (int i = 0; i < 8; ++i) { r[i] = q0[i]; r[i + 8] = q1[i]; }
  return r;
}

__device__ __forceinline__ void store_tile(const v8f& acc, int m0, int n0, int lane,
                                           float* __restrict__ C, long long ldc,
                                           const float* __restrict__ bias,
                                           unsigned short* __restrict__ Cbf, int ldcbf,
                                           int do_tanh) {
  int col = n0 + (lane & 15);
  int rowBase = m0 + ((lane >> 4) << 3);
  float bval = bias ? bias[col] : 0.0f;
#pragma unroll
  for (int r = 0; r < 8; ++r) {
    int row = rowBase + r;
    float v = acc[r] + bval;
    if (do_tanh) v = tanhf(v);
    C[(long long)row * ldc + col] = v;
    if (Cbf) Cbf[(size_t)row * ldcbf + col] = f2bf(v);
  }
}

__global__ void wmma_gemm(const unsigned short* __restrict__ A,
                          const unsigned short* __restrict__ Bt,
                          float* __restrict__ C, long long ldc,
                          const float* __restrict__ bias,
                          unsigned short* __restrict__ Cbf, int ldcbf,
                          int N, int K, int do_tanh) {
  int nBlocks = N >> 5;                                // 2 N-tiles per wave
  int blk = blockIdx.x * (blockDim.x >> 5) + (threadIdx.x >> 5);
  if (blk >= nBlocks) return;                          // whole wave exits together
  int n0 = blk << 5;
  int lane = threadIdx.x & 31;

  v8f acc00 = {}, acc01 = {}, acc10 = {}, acc11 = {};
  for (int k0 = 0; k0 < K; k0 += 32) {
    v16bf a0 = load_frag(A, 0,  K, k0, lane);          // M rows 0..15
    v16bf a1 = load_frag(A, 16, K, k0, lane);          // M rows 16..31
    v16bf b0 = load_frag(Bt, n0,      K, k0, lane);    // N cols n0..n0+15
    v16bf b1 = load_frag(Bt, n0 + 16, K, k0, lane);    // N cols n0+16..n0+31
    acc00 = __builtin_amdgcn_wmma_f32_16x16x32_bf16(false, a0, false, b0, (short)0, acc00, false, false);
    acc01 = __builtin_amdgcn_wmma_f32_16x16x32_bf16(false, a0, false, b1, (short)0, acc01, false, false);
    acc10 = __builtin_amdgcn_wmma_f32_16x16x32_bf16(false, a1, false, b0, (short)0, acc10, false, false);
    acc11 = __builtin_amdgcn_wmma_f32_16x16x32_bf16(false, a1, false, b1, (short)0, acc11, false, false);
  }

  store_tile(acc00, 0,  n0,      lane, C, ldc, bias, Cbf, ldcbf, do_tanh);
  store_tile(acc01, 0,  n0 + 16, lane, C, ldc, bias, Cbf, ldcbf, do_tanh);
  store_tile(acc10, 16, n0,      lane, C, ldc, bias, Cbf, ldcbf, do_tanh);
  store_tile(acc11, 16, n0 + 16, lane, C, ldc, bias, Cbf, ldcbf, do_tanh);
}

// ---------------------------------------------------------------------------
// Per-step kernels
// ---------------------------------------------------------------------------
// xh[b][k] = [ emb[y[b,t]] (512) | ctx_prev (1024) | h_prev (512) ], bf16
__global__ void build_xh(const float* __restrict__ emb, const int* __restrict__ y_train,
                         const float* __restrict__ ctx, const float* __restrict__ h,
                         unsigned short* __restrict__ xh, int t) {
  int idx = blockIdx.x * 256 + threadIdx.x;           // 32*2048
  if (idx >= BB * KGATE) return;
  int b = idx >> 11, k = idx & 2047;
  float v;
  if (k < DD) {
    int tok = y_train[b * TT + t];
    v = emb[(size_t)tok * DD + k];
  } else if (k < 1536) {
    v = ctx[b * 2 * DD + (k - DD)];
  } else {
    v = h[b * DD + (k - 1536)];
  }
  xh[idx] = f2bf(v);
}

// gates [32][2048] (i,f,g,o) -> h,c (in place), + bf16 copies of h
__global__ void lstm_pointwise(const float* __restrict__ gates,
                               float* __restrict__ h, float* __restrict__ c,
                               unsigned short* __restrict__ h_bf,
                               unsigned short* __restrict__ hc_bf) {
  int idx = blockIdx.x * 256 + threadIdx.x;           // 32*512
  if (idx >= BB * DD) return;
  int b = idx / DD, d = idx % DD;
  const float* g = gates + (size_t)b * NGATE;
  float ig = sigmoidf_(g[d]);
  float fg = sigmoidf_(g[DD + d]);
  float gg = tanhf(g[2 * DD + d]);
  float og = sigmoidf_(g[3 * DD + d]);
  float cn = fg * c[idx] + ig * gg;
  float hn = og * tanhf(cn);
  c[idx] = cn;
  h[idx] = hn;
  unsigned short hb = f2bf(hn);
  h_bf[idx] = hb;
  hc_bf[(size_t)b * KC2R + d] = hb;                   // first 512 of [h|ctx]
}

// One block per batch element: scores -> softmax -> ctx; ctx also into hc_bf
__global__ void attention(const float* __restrict__ x_enc,    // [32][64][1024]
                          const float* __restrict__ x_enc_k,  // [32][64][512]
                          const float* __restrict__ proj,     // [32][512]
                          const float* __restrict__ w_att_w,  // [512]
                          const float* __restrict__ w_att_b,  // scalar
                          const unsigned char* __restrict__ x_mask, // [32][64]
                          float* __restrict__ ctx,            // [32][1024]
                          unsigned short* __restrict__ hc_bf) {
  __shared__ float s_sc[LX];
  __shared__ float s_stat[2];
  int b = blockIdx.x;
  int tid = threadIdx.x;

  if (tid < LX) {
    const float* krow = x_enc_k + ((size_t)b * LX + tid) * DD;
    const float* prow = proj + (size_t)b * DD;
    float s = 0.0f;
    for (int d = 0; d < DD; ++d) s += (krow[d] + prow[d]) * w_att_w[d];
    s += w_att_b[0];
    if (x_mask[b * LX + tid]) s = -1.0e9f;
    s_sc[tid] = s;
  }
  __syncthreads();
  if (tid == 0) {
    float mx = s_sc[0];
    for (int l = 1; l < LX; ++l) mx = fmaxf(mx, s_sc[l]);
    float sum = 0.0f;
    for (int l = 0; l < LX; ++l) sum += __expf(s_sc[l] - mx);
    s_stat[0] = mx;
    s_stat[1] = sum;
  }
  __syncthreads();
  if (tid < LX) s_sc[tid] = __expf(s_sc[tid] - s_stat[0]) / s_stat[1];
  __syncthreads();

  // ctx[b][v] = sum_l w[l] * x_enc[b][l][v] ; 256 threads x 4 outputs
  for (int v = tid; v < 2 * DD; v += 256) {
    const float* xrow = x_enc + (size_t)b * LX * 2 * DD + v;
    float acc = 0.0f;
    for (int l = 0; l < LX; ++l) acc += s_sc[l] * xrow[(size_t)l * 2 * DD];
    ctx[b * 2 * DD + v] = acc;
    hc_bf[(size_t)b * KC2R + DD + v] = f2bf(acc);     // last 1024 of [h|ctx]
  }
}

// ---------------------------------------------------------------------------
// Host launcher
// ---------------------------------------------------------------------------
extern "C" void kernel_launch(void* const* d_in, const int* in_sizes, int n_in,
                              void* d_out, int out_size, void* d_ws, size_t ws_size,
                              hipStream_t stream) {
  const float* x_enc      = (const float*)d_in[0];
  const float* x_enc_k    = (const float*)d_in[1];
  const float* dec_init_h = (const float*)d_in[2];
  const float* dec_init_c = (const float*)d_in[3];
  const unsigned char* x_mask = (const unsigned char*)d_in[4];
  const int*   y_train    = (const int*)d_in[5];
  // d_in[6] = y_mask (unused)
  const float* emb        = (const float*)d_in[7];
  const float* W_ih       = (const float*)d_in[8];
  const float* W_hh       = (const float*)d_in[9];
  const float* b_ih       = (const float*)d_in[10];
  const float* b_hh       = (const float*)d_in[11];
  const float* w_trg_w    = (const float*)d_in[12];
  const float* w_trg_b    = (const float*)d_in[13];
  const float* w_att_w    = (const float*)d_in[14];
  const float* w_att_b    = (const float*)d_in[15];
  const float* ctx2read_w = (const float*)d_in[16];
  const float* readout_w  = (const float*)d_in[17];
  float* out = (float*)d_out;

  // ---- workspace partition (aligned to 256B) ----
  char* p = (char*)d_ws;
  auto take = [&](size_t bytes) -> char* {
    char* r = p;
    p += (bytes + 255) & ~(size_t)255;
    return r;
  };
  unsigned short* wcat  = (unsigned short*)take((size_t)NGATE * KGATE * 2); // 8 MB
  unsigned short* wtrg  = (unsigned short*)take((size_t)DD * DD * 2);
  unsigned short* c2r   = (unsigned short*)take((size_t)DD * KC2R * 2);
  unsigned short* rw    = (unsigned short*)take((size_t)VV * DD * 2);       // 32 MB
  float*          biasg = (float*)take(NGATE * 4);
  float*          h     = (float*)take(BB * DD * 4);
  float*          c     = (float*)take(BB * DD * 4);
  float*          ctx   = (float*)take(BB * 2 * DD * 4);
  unsigned short* xh    = (unsigned short*)take((size_t)BB * KGATE * 2);
  float*          gates = (float*)take((size_t)BB * NGATE * 4);
  unsigned short* h_bf  = (unsigned short*)take(BB * DD * 2);
  float*          projb = (float*)take(BB * DD * 4);
  unsigned short* hc_bf = (unsigned short*)take((size_t)BB * KC2R * 2);
  float*          pre_f = (float*)take(BB * DD * 4);
  unsigned short* pre_bf= (unsigned short*)take(BB * DD * 2);

  // ---- one-time weight conversion (runs inside the captured graph; cheap
  //      vs. 90 GFLOP of step work, and keeps the launch deterministic) ----
  cvt_wcat<<<(NGATE * KGATE + 255) / 256, 256, 0, stream>>>(W_ih, W_hh, wcat);
  cvt_direct<<<(DD * DD + 255) / 256, 256, 0, stream>>>(w_trg_w, wtrg, DD * DD);
  cvt_transpose<<<(KC2R * DD + 255) / 256, 256, 0, stream>>>(ctx2read_w, c2r, KC2R, DD);
  cvt_transpose<<<(DD * VV + 255) / 256, 256, 0, stream>>>(readout_w, rw, DD, VV);
  cvt_bias<<<(NGATE + 255) / 256, 256, 0, stream>>>(b_ih, b_hh, biasg, NGATE);
  init_state<<<(BB * 2 * DD + 255) / 256, 256, 0, stream>>>(dec_init_h, dec_init_c, h, c, ctx);

  for (int t = 0; t < TT; ++t) {
    build_xh<<<(BB * KGATE + 255) / 256, 256, 0, stream>>>(emb, y_train, ctx, h, xh, t);

    // gates = xh @ Wcat^T + (b_ih + b_hh): N=2048, K=2048 -> 64 wave-tiles
    wmma_gemm<<<8, 256, 0, stream>>>(xh, wcat, gates, NGATE, biasg,
                                     nullptr, 0, NGATE, KGATE, 0);

    lstm_pointwise<<<(BB * DD + 255) / 256, 256, 0, stream>>>(gates, h, c, h_bf, hc_bf);

    // proj = h @ w_trg_w^T + w_trg_b: N=512, K=512 -> 16 wave-tiles
    wmma_gemm<<<2, 256, 0, stream>>>(h_bf, wtrg, projb, DD, w_trg_b,
                                     nullptr, 0, DD, DD, 0);

    attention<<<BB, 256, 0, stream>>>(x_enc, x_enc_k, projb, w_att_w, w_att_b,
                                      x_mask, ctx, hc_bf);

    // pre_readout = tanh([h|ctx] @ ctx2read_w): N=512, K=1536 -> 16 wave-tiles
    wmma_gemm<<<2, 256, 0, stream>>>(hc_bf, c2r, pre_f, DD, nullptr,
                                     pre_bf, DD, DD, KC2R, 1);

    // logits -> d_out[:, t, :]: N=32000, K=512 -> 1000 wave-tiles
    wmma_gemm<<<125, 256, 0, stream>>>(pre_bf, rw, out + (size_t)t * VV,
                                       (long long)TT * VV, nullptr,
                                       nullptr, 0, VV, DD, 0);
  }
}
